// LinearMixed_67869073211657
// MI455X (gfx1250) — compile-verified, matrix operand order checked by reading, not run on hardware
//
#include <hip/hip_runtime.h>

typedef __attribute__((ext_vector_type(16))) _Float16 v16h;
typedef __attribute__((ext_vector_type(8)))  _Float16 v8h;
typedef __attribute__((ext_vector_type(8)))  float    v8f;
typedef __attribute__((ext_vector_type(4)))  int      v4i;

#define M_TOT 8192
#define N_TOT 4096
#define K_TOT 4096

#define BM 128
#define BN 128
#define BK 64
#define LDSK 72   // padded row stride in halves: 144B -> conflict-free b128 frag loads

// ---- CDNA5 async global->LDS path (guarded; fallback = proven sync path) ----
#if defined(__HIP_DEVICE_COMPILE__) && __has_builtin(__builtin_amdgcn_global_load_async_to_lds_b128)
#define HAVE_ASYNC 1
#else
#define HAVE_ASYNC 0
#endif

#if HAVE_ASYNC
__device__ __forceinline__ void async_copy16(const _Float16* g, _Float16* l) {
  __builtin_amdgcn_global_load_async_to_lds_b128(
      (__attribute__((address_space(1))) v4i*)(g),
      (__attribute__((address_space(3))) v4i*)(l),
      /*offset=*/0, /*cpol=*/0);
}
__device__ __forceinline__ void wait_async0() {
#if __has_builtin(__builtin_amdgcn_s_wait_asynccnt)
  __builtin_amdgcn_s_wait_asynccnt(0);
#else
  asm volatile("s_wait_asynccnt 0" ::: "memory");
#endif
}
#endif

// ---------------- Kernel 1: x f32 -> f16 ----------------
__global__ __launch_bounds__(256) void cvt_x_kernel(const float* __restrict__ x,
                                                    _Float16* __restrict__ xh) {
  size_t idx = ((size_t)blockIdx.x * 256 + threadIdx.x) * 8;
  float4 a = *(const float4*)(x + idx);
  float4 b = *(const float4*)(x + idx + 4);
  v8h h;
  h[0] = (_Float16)a.x; h[1] = (_Float16)a.y; h[2] = (_Float16)a.z; h[3] = (_Float16)a.w;
  h[4] = (_Float16)b.x; h[5] = (_Float16)b.y; h[6] = (_Float16)b.z; h[7] = (_Float16)b.w;
  *(v8h*)(xh + idx) = h;
}

// ---------------- Kernel 2: dequantize 4-bit W -> f16 ----------------
// weight_packed[j] holds one byte: element 2j = low nibble, 2j+1 = high nibble.
// group g = (2j)/256 = j>>7 ; w = q*scale[g] + zero[g]
__global__ __launch_bounds__(256) void deq_w_kernel(const int* __restrict__ wp,
                                                    const float* __restrict__ wpar,
                                                    _Float16* __restrict__ wh) {
  size_t j0 = ((size_t)blockIdx.x * 256 + threadIdx.x) * 4;
  int4 b4 = *(const int4*)(wp + j0);
  size_t g = j0 >> 7;                    // same group for all 4 int32s
  float s = wpar[2 * g];
  float z = wpar[2 * g + 1];
  v8h h;
  int b;
  b = b4.x; h[0] = (_Float16)((float)(b & 15) * s + z); h[1] = (_Float16)((float)((b >> 4) & 15) * s + z);
  b = b4.y; h[2] = (_Float16)((float)(b & 15) * s + z); h[3] = (_Float16)((float)((b >> 4) & 15) * s + z);
  b = b4.z; h[4] = (_Float16)((float)(b & 15) * s + z); h[5] = (_Float16)((float)((b >> 4) & 15) * s + z);
  b = b4.w; h[6] = (_Float16)((float)(b & 15) * s + z); h[7] = (_Float16)((float)((b >> 4) & 15) * s + z);
  *(v8h*)(wh + 2 * j0) = h;
}

// ---------------- Kernel 3: f16 WMMA GEMM: C[M,N] = A[M,K] * B[N,K]^T ----------------
__global__ __launch_bounds__(256) void gemm_f16_wmma(const _Float16* __restrict__ A,
                                                     const _Float16* __restrict__ B,
                                                     float* __restrict__ C) {
  const int tid  = threadIdx.x;
  const int lane = tid & 31;
  const int wave = tid >> 5;
  const int wm   = wave & 3;           // 4 wave-rows of 32
  const int wn   = wave >> 2;          // 2 wave-cols of 64
  const int m0   = blockIdx.y * BM;
  const int n0   = blockIdx.x * BN;

  const int lr = lane & 15;            // row/col within 16x16 fragment
  const int kh = lane >> 4;            // K-half selector per ISA A/B layout

  v8f acc[2][4];
#pragma unroll
  for (int i = 0; i < 2; ++i)
#pragma unroll
    for (int j = 0; j < 4; ++j)
      acc[i][j] = {};

  // staging assignment: 2 threads per row, 32 halves (64 B) each
  const int ldrow = tid >> 1;
  const int ldoff = (tid & 1) * 32;
  const _Float16* gA = A + (size_t)(m0 + ldrow) * K_TOT + ldoff;
  const _Float16* gB = B + (size_t)(n0 + ldrow) * K_TOT + ldoff;

#if HAVE_ASYNC
  // ---- double-buffered LDS, async global->LDS prefetch under WMMA ----
  __shared__ _Float16 sA[2][BM * LDSK];   // 2 x 18 KB
  __shared__ _Float16 sB[2][BN * LDSK];   // 2 x 18 KB
  _Float16* lA[2] = { &sA[0][ldrow * LDSK + ldoff], &sA[1][ldrow * LDSK + ldoff] };
  _Float16* lB[2] = { &sB[0][ldrow * LDSK + ldoff], &sB[1][ldrow * LDSK + ldoff] };
#else
  __shared__ _Float16 sA[1][BM * LDSK];
  __shared__ _Float16 sB[1][BN * LDSK];
  _Float16* lA0 = &sA[0][ldrow * LDSK + ldoff];
  _Float16* lB0 = &sB[0][ldrow * LDSK + ldoff];
#endif

  // one K-stage of WMMAs out of a given pair of LDS tiles
  auto mma_stage = [&](const _Float16* bA, const _Float16* bB) {
#pragma unroll
    for (int kk = 0; kk < BK; kk += 32) {
      v16h af[2], bf[4];
#pragma unroll
      for (int i = 0; i < 2; ++i) {
        const _Float16* p = &bA[(wm * 32 + i * 16 + lr) * LDSK + kk + kh * 8];
        *(uint4*)&af[i]       = *(const uint4*)p;         // K = khalf*8 .. +7
        *((uint4*)&af[i] + 1) = *(const uint4*)(p + 16);  // K = 16 + khalf*8 .. +7
      }
#pragma unroll
      for (int j = 0; j < 4; ++j) {
        const _Float16* p = &bB[(wn * 64 + j * 16 + lr) * LDSK + kk + kh * 8];
        *(uint4*)&bf[j]       = *(const uint4*)p;
        *((uint4*)&bf[j] + 1) = *(const uint4*)(p + 16);
      }
#pragma unroll
      for (int i = 0; i < 2; ++i)
#pragma unroll
        for (int j = 0; j < 4; ++j)
          acc[i][j] = __builtin_amdgcn_wmma_f32_16x16x32_f16(
              false, af[i], false, bf[j], (short)0, acc[i][j], false, false);
    }
  };

#if HAVE_ASYNC
  // prefetch one K-stage (8 x async b128) into buffer `b`
  auto prefetch = [&](int k0, int b) {
#pragma unroll
    for (int u = 0; u < 4; ++u) {
      async_copy16(gA + k0 + u * 8, lA[b] + u * 8);
      async_copy16(gB + k0 + u * 8, lB[b] + u * 8);
    }
  };

  // prologue: stage 0 -> buf0
  prefetch(0, 0);
  wait_async0();
  __syncthreads();

  // pipelined main loop, two stages per iteration with compile-time buffer parity.
  // stages s, s+1 computed; s+1 -> buf1 and s+2 -> buf0 prefetched under the WMMAs.
  for (int k0 = 0; k0 + 2 * BK < K_TOT; k0 += 2 * BK) {
    prefetch(k0 + BK, 1);
    mma_stage(sA[0], sB[0]);
    wait_async0();
    __syncthreads();

    prefetch(k0 + 2 * BK, 0);
    mma_stage(sA[1], sB[1]);
    wait_async0();
    __syncthreads();
  }

  // tail: buf0 holds stage 62; prefetch stage 63 -> buf1
  prefetch(K_TOT - BK, 1);
  mma_stage(sA[0], sB[0]);
  wait_async0();
  __syncthreads();
  mma_stage(sA[1], sB[1]);
#else
  for (int k0 = 0; k0 < K_TOT; k0 += BK) {
#pragma unroll
    for (int u = 0; u < 4; ++u) {
      *(uint4*)(lA0 + u * 8) = *(const uint4*)(gA + k0 + u * 8);
      *(uint4*)(lB0 + u * 8) = *(const uint4*)(gB + k0 + u * 8);
    }
    __syncthreads();
    mma_stage(sA[0], sB[0]);
    __syncthreads();
  }
#endif

  // Epilogue: C VGPR r -> M = base + (lane>=16)*8 + r, N = base + (lane&15)
#pragma unroll
  for (int i = 0; i < 2; ++i) {
#pragma unroll
    for (int j = 0; j < 4; ++j) {
      const int col  = n0 + wn * 64 + j * 16 + lr;
      const int rowb = m0 + wm * 32 + i * 16 + kh * 8;
      float* outp = C + (size_t)rowb * N_TOT + col;
#pragma unroll
      for (int r = 0; r < 8; ++r)
        outp[(size_t)r * N_TOT] = acc[i][j][r];
    }
  }
}

extern "C" void kernel_launch(void* const* d_in, const int* in_sizes, int n_in,
                              void* d_out, int out_size, void* d_ws, size_t ws_size,
                              hipStream_t stream) {
  const float* x     = (const float*)d_in[0];
  const int*   wp    = (const int*)d_in[1];
  const float* wpar  = (const float*)d_in[2];
  float*       out   = (float*)d_out;

  _Float16* xh = (_Float16*)d_ws;                                      // 64 MB
  _Float16* wh = (_Float16*)((char*)d_ws + (size_t)M_TOT * K_TOT * 2); // +32 MB

  cvt_x_kernel<<<(M_TOT * (size_t)K_TOT) / (256 * 8), 256, 0, stream>>>(x, xh);
  deq_w_kernel<<<((size_t)N_TOT * K_TOT / 2) / (256 * 4), 256, 0, stream>>>(wp, wpar, wh);

  dim3 grid(N_TOT / BN, M_TOT / BM);   // 32 x 64
  gemm_f16_wmma<<<grid, 256, 0, stream>>>(xh, wh, out);
}
